// Attention_2233382994505
// MI455X (gfx1250) — matrix operand order repaired and checked by reading, not compiled, not run on hardware
//
#include <hip/hip_runtime.h>

// ---------------------------------------------------------------------------
// Types (plain ext vectors: trivially constructible -> safe in unions)
// ---------------------------------------------------------------------------
typedef __attribute__((ext_vector_type(8)))  float          v8f;
typedef __attribute__((ext_vector_type(16))) __bf16         v16bf;
typedef __attribute__((ext_vector_type(4)))  unsigned int   uint4v;
typedef __attribute__((ext_vector_type(4)))  float          float4v;
typedef __attribute__((ext_vector_type(4)))  unsigned short ushort4v;
typedef __attribute__((ext_vector_type(4)))  unsigned int   u32x4;
typedef __attribute__((ext_vector_type(8)))  int            i32x8;
typedef __attribute__((ext_vector_type(4)))  int            i32x4;

#define LDT   40   // LDS row stride (elements) for 32-wide tiles (bank-conflict-free)
#define LDK   72   // LDS row stride for 64-wide feature tiles

static __device__ __forceinline__ unsigned short f2bf(float f) {
  union { float f; unsigned int u; } x; x.f = f;
  unsigned int r = x.u + 0x7FFFu + ((x.u >> 16) & 1u);   // round-to-nearest-even
  return (unsigned short)(r >> 16);
}

static __device__ __forceinline__ v8f wmma_bf16(v16bf a, v16bf b, v8f c) {
  return __builtin_amdgcn_wmma_f32_16x16x32_bf16(false, a, false, b, (short)0, c,
                                                 false, false);
}

// ---------------------------------------------------------------------------
// CDNA5 async / tensor data movement
// ---------------------------------------------------------------------------
// Async global->LDS 128b load (GV mode, per-lane addresses), tracked by ASYNCcnt.
static __device__ __forceinline__ void async_ld128(unsigned ldsOff, const void* g) {
  asm volatile("global_load_async_to_lds_b128 %0, %1, off"
               :: "v"(ldsOff), "v"(g) : "memory");
}
static __device__ __forceinline__ void wait_async0() {
  asm volatile("s_wait_asynccnt 0x0" ::: "memory");
}

// TDM 2-D tile load: tensor dims == tile dims (no OOB), bf16 elements,
// LDS padding of padAmount(code) after every padInterval(code) chunk.
// padInterval code: bytes = 8 << code ; padAmount code: dwords = code+1.
static __device__ __forceinline__ void tdm_load_2d(unsigned ldsOff, const void* gaddr,
                                                   int tileW, int tileH,
                                                   long long strideElems,
                                                   int padInterval, int padAmount) {
  unsigned long long ga = (unsigned long long)gaddr;
  u32x4 g0;
  g0[0] = 1u;                                               // count=1, user mode
  g0[1] = ldsOff;                                           // lds_addr (bytes)
  g0[2] = (unsigned)ga;                                     // global_addr[31:0]
  g0[3] = (unsigned)((ga >> 32) & 0x01FFFFFFull) | (2u << 30); // addr[56:32] | type=2
  i32x8 g1;
  g1[0] = (1 << 16) | (1 << 20) | (padInterval << 22) | (padAmount << 25); // 2B, pad_en
  g1[1] = (tileW & 0xFFFF) << 16;                           // tensor_dim0 lo16
  g1[2] = (((unsigned)tileW) >> 16) | ((tileH & 0xFFFF) << 16); // dim0 hi | dim1 lo
  g1[3] = (((unsigned)tileH) >> 16) | ((tileW & 0xFFFF) << 16); // dim1 hi | tile_dim0
  g1[4] = (tileH & 0xFFFF);                                 // tile_dim1 ; tile_dim2=0
  g1[5] = (int)(strideElems & 0xFFFFFFFFll);                // dim0_stride lo32
  g1[6] = (int)((strideElems >> 32) & 0xFFFFll);            // dim0_stride hi16
  g1[7] = 0;
  i32x4 z4 = {0, 0, 0, 0};
  i32x8 z8 = {0, 0, 0, 0, 0, 0, 0, 0};
  __builtin_amdgcn_tensor_load_to_lds(g0, g1, z4, z4, z8, 0);
}

// ---------------------------------------------------------------------------
// WMMA fragment loads
// ---------------------------------------------------------------------------
// A-operand (16x32 bf16): lane L -> row (L&15); K chunks {k0..k0+7, k0+16..k0+23},
// k0 = 0 (lanes 0-15) or 8 (lanes 16-31).
static __device__ __forceinline__ v16bf load_frag_a(const unsigned short* base, int stride,
                                                    int row0, int col0, int lane) {
  int r = row0 + (lane & 15);
  int c = col0 + ((lane & 16) ? 8 : 0);
  const unsigned short* p = base + r * stride + c;
  union { v16bf v; uint4v u[2]; } f;
  f.u[0] = *(const uint4v*)(p);
  f.u[1] = *(const uint4v*)(p + 16);
  return f.v;
}

// B-operand (32x16 bf16) from a transposed ([N][K]) tile:
// lane L -> column (L&15); K chunks {k0..k0+7, k0+16..k0+23}, k0 = 0 (lanes 0-15) or 8 (lanes 16-31).
static __device__ __forceinline__ v16bf load_frag_b(const unsigned short* base, int stride,
                                                    int row0, int col0, int lane) {
  int r = row0 + (lane & 15);
  int c = col0 + ((lane & 16) ? 8 : 0);
  const unsigned short* p = base + r * stride + c;
  union { v16bf v; uint4v u[2]; } f;
  f.u[0] = *(const uint4v*)(p);
  f.u[1] = *(const uint4v*)(p + 16);
  return f.v;
}

// 16-lane xor reductions via ds_swizzle (masks < 16 keep lane halves separate)
template <int MASK>
static __device__ __forceinline__ float swz_xor(float v) {
  return __int_as_float(
      __builtin_amdgcn_ds_swizzle(__float_as_int(v), (MASK << 10) | 0x1f));
}
static __device__ __forceinline__ float rmax16(float v) {
  v = fmaxf(v, swz_xor<1>(v)); v = fmaxf(v, swz_xor<2>(v));
  v = fmaxf(v, swz_xor<4>(v)); v = fmaxf(v, swz_xor<8>(v));
  return v;
}
static __device__ __forceinline__ float rsum16(float v) {
  v += swz_xor<1>(v); v += swz_xor<2>(v);
  v += swz_xor<4>(v); v += swz_xor<8>(v);
  return v;
}

// ---------------------------------------------------------------------------
// Precision-conversion kernels
// ---------------------------------------------------------------------------
__global__ void __launch_bounds__(256)
k_f32_to_bf16(const float* __restrict__ in, unsigned short* __restrict__ out, int n4) {
  int i = blockIdx.x * 256 + threadIdx.x;
  if (i >= n4) return;
  float4v a = ((const float4v*)in)[i];
  ushort4v o;
  o[0] = f2bf(a[0]); o[1] = f2bf(a[1]); o[2] = f2bf(a[2]); o[3] = f2bf(a[3]);
  ((ushort4v*)out)[i] = o;
}

// out[n*K + k] = bf16(in[k*Nn + n])   (writes coalesced)
__global__ void __launch_bounds__(256)
k_transpose_bf16(const float* __restrict__ in, unsigned short* __restrict__ out,
                 int K, int Nn) {
  long long idx = (long long)blockIdx.x * 256 + threadIdx.x;
  if (idx >= (long long)K * Nn) return;
  int n = (int)(idx / K), k = (int)(idx % K);
  out[idx] = f2bf(in[(size_t)k * Nn + n]);
}

// ---------------------------------------------------------------------------
// GEMM mainloop: C[128x128] = A[128x1024] * Bt[128x1024]^T  (bf16, [row][K])
// 8 waves in 2x4; each wave 64x32 -> acc[4][2]. Async-LDS double-buffered.
// ---------------------------------------------------------------------------
static __device__ __forceinline__ void
gemm_mainloop(const unsigned short* __restrict__ A, const unsigned short* __restrict__ Bt,
              int mBase, int nBase, unsigned short* ldsA, unsigned short* ldsB,
              v8f acc[4][2]) {
  const int tid = threadIdx.x, lane = tid & 31, w = tid >> 5;
  const int wm = w >> 2, wn = w & 3;
  const int row = tid >> 1, half = tid & 1;
  const unsigned short* ga = A + (size_t)(mBase + row) * 1024 + half * 16;
  const unsigned short* gb = Bt + (size_t)(nBase + row) * 1024 + half * 16;
  const unsigned la = (unsigned)(size_t)(ldsA + row * LDT + half * 16);
  const unsigned lb = (unsigned)(size_t)(ldsB + row * LDT + half * 16);
  const unsigned bufB = 128 * LDT * 2;   // buffer stride in bytes

  // prologue: stage kb=0 into buffer 0
  async_ld128(la, ga);      async_ld128(la + 16, ga + 8);
  async_ld128(lb, gb);      async_ld128(lb + 16, gb + 8);

  for (int kb = 0; kb < 32; ++kb) {
    const int cur = kb & 1;
    wait_async0();        // this wave's loads for buffer `cur` are in LDS
    __syncthreads();      // everyone's loads landed; previous reads retired
    if (kb + 1 < 32) {    // prefetch next k-slab into the other buffer
      const unsigned short* na = ga + (kb + 1) * 32;
      const unsigned short* nb = gb + (kb + 1) * 32;
      const unsigned da = la + (cur ^ 1) * bufB, db = lb + (cur ^ 1) * bufB;
      async_ld128(da, na);  async_ld128(da + 16, na + 8);
      async_ld128(db, nb);  async_ld128(db + 16, nb + 8);
    }
    const unsigned short* tA = ldsA + cur * (128 * LDT);
    const unsigned short* tB = ldsB + cur * (128 * LDT);
    v16bf af[4], bfr[2];
#pragma unroll
    for (int i = 0; i < 4; ++i) af[i]  = load_frag_a(tA, LDT, wm * 64 + 16 * i, 0, lane);
#pragma unroll
    for (int j = 0; j < 2; ++j) bfr[j] = load_frag_b(tB, LDT, wn * 32 + 16 * j, 0, lane);
#pragma unroll
    for (int i = 0; i < 4; ++i)
#pragma unroll
      for (int j = 0; j < 2; ++j)
        acc[i][j] = wmma_bf16(af[i], bfr[j], acc[i][j]);
  }
}

// QKV GEMM: M=4096 (b*2048+n), N=3072 (t*1024 + h*64 + d). Scatter to Q/K/Vt (bf16).
__global__ void __launch_bounds__(256)
k_gemm_qkv(const unsigned short* __restrict__ A, const unsigned short* __restrict__ Bt,
           const float* __restrict__ bias, unsigned short* __restrict__ Qo,
           unsigned short* __restrict__ Ko, unsigned short* __restrict__ Vto) {
  __shared__ unsigned short ldsA[2 * 128 * LDT];
  __shared__ unsigned short ldsB[2 * 128 * LDT];
  const int tid = threadIdx.x, lane = tid & 31, w = tid >> 5;
  const int wm = w >> 2, wn = w & 3;
  const int mBase = blockIdx.x * 128, nBase = blockIdx.y * 128;
  v8f acc[4][2] = {};
  gemm_mainloop(A, Bt, mBase, nBase, ldsA, ldsB, acc);

  const int rowOff = ((lane >> 4) & 1) * 8, colOff = lane & 15;
  const int t = nBase >> 10;                     // block-uniform: which of q/k/v
  float bj[2]; int hh[2], dd[2];
#pragma unroll
  for (int j = 0; j < 2; ++j) {
    int gn = nBase + wn * 32 + 16 * j + colOff;
    bj[j] = bias[gn];
    hh[j] = (gn & 1023) >> 6;
    dd[j] = gn & 63;
  }
  if (t < 2) {
    unsigned short* dst = (t == 0) ? Qo : Ko;    // [b,h,n,64]
#pragma unroll
    for (int j = 0; j < 2; ++j)
#pragma unroll
      for (int i = 0; i < 4; ++i)
#pragma unroll
        for (int r = 0; r < 8; ++r) {
          int gm = mBase + wm * 64 + 16 * i + r + rowOff;
          int bb = gm >> 11, nn = gm & 2047;
          dst[(((size_t)(bb * 16 + hh[j])) * 2048 + nn) * 64 + dd[j]] =
              f2bf(acc[i][j][r] + bj[j]);
        }
  } else {                                       // V transposed: [b,h,64,n]
#pragma unroll
    for (int j = 0; j < 2; ++j)
#pragma unroll
      for (int i = 0; i < 4; ++i)
#pragma unroll
        for (int r = 0; r < 8; ++r) {
          int gm = mBase + wm * 64 + 16 * i + r + rowOff;
          int bb = gm >> 11, nn = gm & 2047;
          Vto[(((size_t)(bb * 16 + hh[j])) * 64 + dd[j]) * 2048 + nn] =
              f2bf(acc[i][j][r] + bj[j]);
        }
  }
}

// Projection GEMM: M=4096, N=1024, f32 output + bias.
__global__ void __launch_bounds__(256)
k_gemm_proj(const unsigned short* __restrict__ A, const unsigned short* __restrict__ Bt,
            const float* __restrict__ bias, float* __restrict__ out) {
  __shared__ unsigned short ldsA[2 * 128 * LDT];
  __shared__ unsigned short ldsB[2 * 128 * LDT];
  const int tid = threadIdx.x, lane = tid & 31, w = tid >> 5;
  const int wm = w >> 2, wn = w & 3;
  const int mBase = blockIdx.x * 128, nBase = blockIdx.y * 128;
  v8f acc[4][2] = {};
  gemm_mainloop(A, Bt, mBase, nBase, ldsA, ldsB, acc);

  const int rowOff = ((lane >> 4) & 1) * 8, colOff = lane & 15;
#pragma unroll
  for (int j = 0; j < 2; ++j) {
    int gn = nBase + wn * 32 + 16 * j + colOff;
    float bj = bias[gn];
#pragma unroll
    for (int i = 0; i < 4; ++i)
#pragma unroll
      for (int r = 0; r < 8; ++r) {
        int gm = mBase + wm * 64 + 16 * i + r + rowOff;
        out[(size_t)gm * 1024 + gn] = acc[i][j][r] + bj;
      }
  }
}

// ---------------------------------------------------------------------------
// Flash attention: grid (N/128, H, B); block 256 = 8 waves, 16 queries/wave.
// Q,K: [b,h,n,64] bf16;  Vt: [b,h,64,n] bf16;  O: [b,n,h*64+d] bf16.
// K/V^T tiles staged by the Tensor Data Mover (wave 0), double-buffered.
// ---------------------------------------------------------------------------
__global__ void __launch_bounds__(256)
k_attn(const unsigned short* __restrict__ Q, const unsigned short* __restrict__ K,
       const unsigned short* __restrict__ Vt, unsigned short* __restrict__ O) {
  __shared__ unsigned short ldsK[2][32 * LDK];     // 32 keys x 64 feats (padded 72)
  __shared__ unsigned short ldsV[2][64 * LDT];     // 64 feats x 32 keys (padded 40)
  __shared__ unsigned short ldsP[8 * 16 * LDT];    // per-wave P tiles (16 q x 32 keys)
  const int tid = threadIdx.x, lane = tid & 31, w = tid >> 5;
  const int h = blockIdx.y, b = blockIdx.z;
  const int qbase = blockIdx.x * 128 + w * 16;
  const size_t head = ((size_t)(b * 16 + h)) * 2048 * 64;
  const unsigned short* Qh = Q + head;
  const unsigned short* Kh = K + head;
  const unsigned short* Vh = Vt + head;
  const int r16 = lane & 15;
  const int kb0 = (lane & 16) ? 8 : 0;

  // Preload Q fragments (A-layout) straight from global
  v16bf aq[2];
#pragma unroll
  for (int s = 0; s < 2; ++s) {
    const unsigned short* p = Qh + (size_t)(qbase + r16) * 64 + s * 32 + kb0;
    union { v16bf v; uint4v u[2]; } f;
    f.u[0] = *(const uint4v*)p;
    f.u[1] = *(const uint4v*)(p + 16);
    aq[s] = f.v;
  }

  v8f o[4] = {};
  float mi[8], li[8];
#pragma unroll
  for (int r = 0; r < 8; ++r) { mi[r] = -1e30f; li[r] = 0.f; }
  unsigned short* ldsPw = ldsP + w * (16 * LDT);

  if (w == 0) {   // stage tiles for t=0 into buffer 0
    tdm_load_2d((unsigned)(size_t)&ldsK[0][0], Kh, 64, 32, 64, 4, 3);
    tdm_load_2d((unsigned)(size_t)&ldsV[0][0], Vh, 32, 64, 2048, 3, 3);
  }

  for (int t = 0; t < 64; ++t) {
    const int cur = t & 1;
    if (w == 0) __builtin_amdgcn_s_wait_tensorcnt(0);
    __syncthreads();                    // tiles for `cur` visible to all waves
    if (w == 0 && t + 1 < 64) {         // TDM-prefetch next key block
      tdm_load_2d((unsigned)(size_t)&ldsK[cur ^ 1][0],
                  Kh + (size_t)(t + 1) * 32 * 64, 64, 32, 64, 4, 3);
      tdm_load_2d((unsigned)(size_t)&ldsV[cur ^ 1][0],
                  Vh + (size_t)(t + 1) * 32, 32, 64, 2048, 3, 3);
    }

    // S = Q K^T  (two 16-key tiles, K-dim 64 = 2 steps)
    v8f s0 = {}, s1 = {};
#pragma unroll
    for (int ss = 0; ss < 2; ++ss) {
      v16bf b0 = load_frag_b(ldsK[cur], LDK, 0,  ss * 32, lane);
      v16bf b1 = load_frag_b(ldsK[cur], LDK, 16, ss * 32, lane);
      s0 = wmma_bf16(aq[ss], b0, s0);
      s1 = wmma_bf16(aq[ss], b1, s1);
    }

    // Online softmax per row (rows r and r+8 split by lane half)
#pragma unroll
    for (int r = 0; r < 8; ++r) {
      float a0 = s0[r] * 0.125f, a1 = s1[r] * 0.125f;
      float mnew = fmaxf(mi[r], rmax16(fmaxf(a0, a1)));
      float fac = exp2f((mi[r] - mnew) * 1.44269504f);
      mi[r] = mnew;
      float p0 = exp2f((a0 - mnew) * 1.44269504f);
      float p1 = exp2f((a1 - mnew) * 1.44269504f);
      li[r] = li[r] * fac + rsum16(p0 + p1);
      int prow = r + ((lane >> 4) & 1) * 8;
      ldsPw[prow * LDT + r16]      = f2bf(p0);
      ldsPw[prow * LDT + 16 + r16] = f2bf(p1);
#pragma unroll
      for (int j = 0; j < 4; ++j) o[j][r] *= fac;
    }

    // O += P V  (wave-private P: same-wave DS ops are in-order, no barrier)
    v16bf ap = load_frag_a(ldsPw, LDT, 0, 0, lane);
#pragma unroll
    for (int j = 0; j < 4; ++j) {
      v16bf bv = load_frag_b(ldsV[cur], LDT, 16 * j, 0, lane);
      o[j] = wmma_bf16(ap, bv, o[j]);
    }
  }

  const int rowOff = ((lane >> 4) & 1) * 8;
#pragma unroll
  for (int r = 0; r < 8; ++r) {
    float inv = 1.0f / li[r];
    int q = qbase + r + rowOff;
    size_t base = ((size_t)(b * 2048 + q)) * 1024 + h * 64 + r16;
#pragma unroll
    for (int j = 0; j < 4; ++j) O[base + 16 * j] = f2bf(o[j][r] * inv);
  }
}

// ---------------------------------------------------------------------------
// Launch
// ---------------------------------------------------------------------------
extern "C" void kernel_launch(void* const* d_in, const int* in_sizes, int n_in,
                              void* d_out, int out_size, void* d_ws, size_t ws_size,
                              hipStream_t stream) {
  const float* x      = (const float*)d_in[0];
  const float* W_qkv  = (const float*)d_in[1];
  const float* b_qkv  = (const float*)d_in[2];
  const float* W_proj = (const float*)d_in[3];
  const float* b_proj = (const float*)d_in[4];
  float* out = (float*)d_out;

  char* p = (char*)d_ws;
  unsigned short* xb     = (unsigned short*)p; p += 4096ull * 1024 * 2;   // x bf16
  unsigned short* wqkvT  = (unsigned short*)p; p += 3072ull * 1024 * 2;   // W_qkv^T bf16
  unsigned short* wprojT = (unsigned short*)p; p += 1024ull * 1024 * 2;   // W_proj^T bf16
  unsigned short* Qb     = (unsigned short*)p; p += 2ull * 16 * 2048 * 64 * 2;
  unsigned short* Kb     = (unsigned short*)p; p += 2ull * 16 * 2048 * 64 * 2;
  unsigned short* Vtb    = (unsigned short*)p; p += 2ull * 16 * 2048 * 64 * 2;
  unsigned short* AO     = (unsigned short*)p; p += 4096ull * 1024 * 2;   // attn out bf16

  k_f32_to_bf16<<<4096, 256, 0, stream>>>(x, xb, 4096 * 1024 / 4);
  k_transpose_bf16<<<(1024 * 3072 + 255) / 256, 256, 0, stream>>>(W_qkv, wqkvT, 1024, 3072);
  k_transpose_bf16<<<(1024 * 1024 + 255) / 256, 256, 0, stream>>>(W_proj, wprojT, 1024, 1024);

  k_gemm_qkv<<<dim3(32, 24), 256, 0, stream>>>(xb, wqkvT, b_qkv, Qb, Kb, Vtb);
  k_attn<<<dim3(16, 16, 2), 256, 0, stream>>>(Qb, Kb, Vtb, AO);
  k_gemm_proj<<<dim3(32, 8), 256, 0, stream>>>(AO, wprojT, b_proj, out);
}
